// GraphSAGE_24172075942153
// MI455X (gfx1250) — matrix operand order repaired and checked by reading, not compiled, not run on hardware
//
#include <hip/hip_runtime.h>
#include <hip/hip_bf16.h>

typedef __attribute__((ext_vector_type(16))) _Float16 v16h;
typedef __attribute__((ext_vector_type(8)))  float    v8f;

#define GN 16384
#define GD 32
#define KTILES (GN / 32)      // 512
#define WAVES 8               // waves per workgroup (wave32)

// ---------------------------------------------------------------------------
// Prep: repack h (N x 32 fp32, row-major) into f16 arranged as WMMA B operands.
// B tile (32x16 f16, v_wmma_f32_16x16x32_f16): lanes 0-15 hold K=kt*32+{0..7,16..23}
// (col = lane), lanes 16-31 hold K=kt*32+{8..15,24..31} (col = lane-16); 16 f16/lane.
// hB index: (((kt*2 + half)*32 + lane)*16 + j), half selects D cols 0-15/16-31.
// ---------------------------------------------------------------------------
__global__ __launch_bounds__(256) void sage_prep_hB(const float* __restrict__ h,
                                                    _Float16* __restrict__ hB) {
  int idx  = blockIdx.x * 256 + threadIdx.x;      // 512*2*32*16 = 524288 total
  int j    = idx & 15;
  int lane = (idx >> 4) & 31;
  int half = (idx >> 9) & 1;
  int kt   = idx >> 10;
  int k    = kt * 32 + ((lane >> 4) << 3) + (j & 7) + ((j >> 3) << 4);    // +8 for upper lane half
  int col  = half * 16 + (lane & 15);
  hB[idx] = (_Float16)h[k * GD + col];
}

// ---------------------------------------------------------------------------
// Main: one wave per 16-row tile. Stream A once (1 GiB -> HBM-bound), WMMA the
// binary mask against repacked h. Neighbor counts come from a third WMMA
// against an all-ones B: in C-layout, acc_c[j] is the row count for the same
// (lane, j) slot as acc0[j]/acc1[j], so no cross-lane reduction is needed.
// Then agg = (acc + h) / (cnt + 1), out = leaky_relu(agg @ W^T).
// ---------------------------------------------------------------------------
__global__ __launch_bounds__(256) void sage_main(const float* __restrict__ A,
                                                 const float* __restrict__ h,
                                                 const float* __restrict__ W,
                                                 const _Float16* __restrict__ hB,
                                                 float* __restrict__ out) {
  __shared__ float sWt[GD * GD];           // W transposed: sWt[d*32 + c] = W[c][d]
  __shared__ float sAgg[WAVES][16][GD];    // per-wave agg tile staging

  const int tid  = threadIdx.x;
  const int wave = tid >> 5;
  const int lane = tid & 31;
  const int halfSel = lane >> 4;           // 0: K 0-7/16-23, 1: K 8-15/24-31

  // Load weight transposed (conflict-free epilogue reads).
  for (int i = tid; i < GD * GD; i += 256) {
    int c = i >> 5, d = i & 31;
    sWt[d * GD + c] = W[i];
  }

  const size_t rowBase = ((size_t)blockIdx.x * WAVES + wave) * 16;
  const float* aRow = A + (rowBase + (lane & 15)) * (size_t)GN;
  const int run0 = halfSel * 8;            // first contiguous 8-float run

  v16h ones;
#pragma unroll
  for (int j = 0; j < 16; ++j) ones[j] = (_Float16)1.0f;

  v8f acc0 = {};                           // mask@h, output cols 0-15
  v8f acc1 = {};                           // mask@h, output cols 16-31
  v8f accC = {};                           // mask@ones = neighbor counts

#pragma unroll 2
  for (int kt = 0; kt < KTILES; ++kt) {
    const int kBase = kt * 32;
    // A-operand layout (16-bit A 16x32): lane<16 -> K {0..7, 16..23},
    // lane>=16 -> K {8..15, 24..31}; two contiguous float4-pair runs.
    const float4 a0 = *(const float4*)(aRow + kBase + run0);
    const float4 a1 = *(const float4*)(aRow + kBase + run0 + 4);
    const float4 a2 = *(const float4*)(aRow + kBase + run0 + 16);
    const float4 a3 = *(const float4*)(aRow + kBase + run0 + 20);

    __builtin_prefetch(aRow + kBase + 32 * 8, 0, 1);   // ~1 KB/row ahead

    float av[16] = {a0.x, a0.y, a0.z, a0.w, a1.x, a1.y, a1.z, a1.w,
                    a2.x, a2.y, a2.z, a2.w, a3.x, a3.y, a3.z, a3.w};
    v16h af;
#pragma unroll
    for (int j = 0; j < 16; ++j)
      af[j] = (av[j] != 0.0f) ? (_Float16)1.0f : (_Float16)0.0f;

    const v16h b0 = *(const v16h*)(hB + (((size_t)kt * 2 + 0) * 32 + lane) * 16);
    const v16h b1 = *(const v16h*)(hB + (((size_t)kt * 2 + 1) * 32 + lane) * 16);

    acc0 = __builtin_amdgcn_wmma_f32_16x16x32_f16(false, af, false, b0,
                                                  (short)0, acc0, false, false);
    acc1 = __builtin_amdgcn_wmma_f32_16x16x32_f16(false, af, false, b1,
                                                  (short)0, acc1, false, false);
    accC = __builtin_amdgcn_wmma_f32_16x16x32_f16(false, af, false, ones,
                                                  (short)0, accC, false, false);
  }

  // C/D layout: element j -> row = j + 8*halfSel, col = (lane&15) [+16 acc1].
  // accC[j] holds rowsum(mask) for that same row (replicated across cols).
#pragma unroll
  for (int j = 0; j < 8; ++j) {
    const int r     = j + 8 * halfSel;
    const float inv = 1.0f / (accC[j] + 1.0f);
    const size_t row = rowBase + r;
    const int c0 = lane & 15;
    sAgg[wave][r][c0]      = (acc0[j] + h[row * GD + c0])      * inv;
    sAgg[wave][r][c0 + 16] = (acc1[j] + h[row * GD + c0 + 16]) * inv;
  }
  __syncthreads();

  // out[row][c] = leaky_relu( sum_d agg[row][d] * W[c][d] )  (fp32 VALU; tiny)
  const int c = lane;
#pragma unroll 4
  for (int r = 0; r < 16; ++r) {
    float s = 0.0f;
#pragma unroll
    for (int d = 0; d < GD; ++d)
      s = fmaf(sAgg[wave][r][d], sWt[d * GD + c], s);
    out[(rowBase + r) * GD + c] = (s > 0.0f) ? s : 0.01f * s;
  }
}

extern "C" void kernel_launch(void* const* d_in, const int* in_sizes, int n_in,
                              void* d_out, int out_size, void* d_ws, size_t ws_size,
                              hipStream_t stream) {
  const float* A = (const float*)d_in[0];   // [N, N] fp32 (dense 0/x adjacency)
  const float* h = (const float*)d_in[1];   // [N, 32] fp32
  const float* W = (const float*)d_in[2];   // [32, 32] fp32
  _Float16* hB = (_Float16*)d_ws;           // 512*2*32*16 f16 = 1 MB

  sage_prep_hB<<<2048, 256, 0, stream>>>(h, hB);
  sage_main<<<GN / (16 * WAVES), 256, 0, stream>>>(A, h, W, hB, (float*)d_out);
}
